// Net_6768868458782
// MI455X (gfx1250) — matrix operand order repaired and checked by reading, not compiled, not run on hardware
//
#include <hip/hip_runtime.h>
#include <hip/hip_bf16.h>
#include <math.h>

// ---------------------------------------------------------------------------
// Problem constants (match the reference)
// ---------------------------------------------------------------------------
#define Bc   256
#define Nc   1024
#define Ec   (256 * 1024 * 16)     // 4,194,304 edges
#define NH   128                   // NHID == F_IN
#define K1c  512
#define K2c  256
#define K3c  128
#define NODES1 (Bc * Nc)           // 262144
#define NODES2 (Bc * K1c)          // 131072
#define NODES3 (Bc * K2c)          // 65536

typedef __bf16 bf16_t;
typedef __attribute__((ext_vector_type(16))) __bf16 v16bf;
typedef __attribute__((ext_vector_type(8)))  __bf16 v8bf;
typedef __attribute__((ext_vector_type(8)))  float  v8f;

// ---------------------------------------------------------------------------
// One-time per launch: W[k][n] (f32 row-major) -> WT[n][k] (bf16 col-major)
// so each WMMA lane's B fragment is 16 contiguous bf16 (2x global_load_b128).
// ---------------------------------------------------------------------------
__global__ void transpose_w_bf16_kernel(const float* __restrict__ W,
                                        bf16_t* __restrict__ WT)
{
    int i = blockIdx.x * blockDim.x + threadIdx.x;   // 0 .. 16383
    int n = i >> 7, k = i & 127;
    WT[n * NH + k] = (bf16_t)W[k * NH + n];
}

// ---------------------------------------------------------------------------
// Fused (optionally gathered+gated) GEMM + bias + ReLU via bf16 WMMA.
//   Y[m, :] = relu( (perm ? tanh(gs[perm[m]]) * X[perm[m], :] : X[m, :]) @ W + b )
// Block = 256 threads = 8 waves, tile = 64 rows x 128 cols.
// Each wave owns one 16-wide N tile and iterates 4 M-subtiles, reusing each
// B fragment across 4 WMMA ops. A staged in LDS as bf16 (16 KB).
// WT is bf16 column-major (see transpose kernel). M must be a multiple of 64.
// ---------------------------------------------------------------------------
__global__ __launch_bounds__(256)
void gemm_relu_wmma(const float* __restrict__ X, const bf16_t* __restrict__ WT,
                    const float* __restrict__ bias, float* __restrict__ Y,
                    const int* __restrict__ perm, const float* __restrict__ gs)
{
    __shared__ __align__(32) bf16_t tile[64 * NH];   // 64 x 128 bf16

    const int tid = threadIdx.x;
    const int m0  = blockIdx.x * 64;

    // --- stage A tile: 64x128 f32 -> bf16, 8 contiguous elems x 4 rows/thread
    {
        const int colq = (tid & 15) * 8;
        #pragma unroll
        for (int r = 0; r < 4; ++r) {
            const int row  = (tid >> 4) + r * 16;
            const int grow = m0 + row;
            const int node = perm ? perm[grow] : grow;
            const float scale = gs ? tanhf(gs[node]) : 1.0f;
            const float* sp = X + (size_t)node * NH + colq;
            #pragma unroll
            for (int j = 0; j < 8; ++j)
                tile[row * NH + colq + j] = (bf16_t)(sp[j] * scale);
        }
    }
    __syncthreads();

    const int wave = tid >> 5;            // 0..7 -> N tile
    const int lane = tid & 31;
    const int half = lane >> 4;           // lane group 0/1
    const int lrow = lane & 15;
    const int nc   = wave * 16 + lrow;    // output column (B/C/D lane mapping)

    v8f c[4] = {v8f{}, v8f{}, v8f{}, v8f{}};

    #pragma unroll
    for (int kk = 0; kk < 4; ++kk) {      // K = 128 in 4 steps of 32
        // B 32x16 fragment: lane = column; lanes 0-15 K=0..15, 16-31 K=16..31.
        // WT col-major => 16 contiguous bf16, 32B-aligned.
        const int koff = kk * 32 + half * 16;
        const v16bf b = *(const v16bf*)(WT + (size_t)nc * NH + koff);

        // A 16x32 layout (matches B): lanes 0-15 K 0-15,
        // lanes 16-31 K 16-31 -> two 16B ds_load_b128 per fragment.
        const int kb = kk * 32 + half * 16;
        #pragma unroll
        for (int mt = 0; mt < 4; ++mt) {
            const int row = mt * 16 + lrow;
            union { v16bf v; struct { v8bf lo, hi; } s; } au;
            au.s.lo = *(const v8bf*)(tile + row * NH + kb);
            au.s.hi = *(const v8bf*)(tile + row * NH + kb + 8);
            c[mt] = __builtin_amdgcn_wmma_f32_16x16x32_bf16(
                        false, au.v, false, b, (short)0, c[mt], false, false);
        }
    }

    // C/D layout: VGPR r -> M = r (lanes 0-15) / r+8 (lanes 16-31), N = lane&15
    const float bn = bias[nc];
    #pragma unroll
    for (int mt = 0; mt < 4; ++mt) {
        const int mbase = m0 + mt * 16 + half * 8;
        #pragma unroll
        for (int r = 0; r < 8; ++r) {
            float v = c[mt][r] + bn;
            Y[(size_t)(mbase + r) * NH + nc] = v > 0.0f ? v : 0.0f;
        }
    }
}

// ---------------------------------------------------------------------------
// hs[i] = dot(H[i, 0:128], Ws)  (GCN score linear, no bias here)
// One wave32 per node, 4 contiguous f32 per lane, shuffle reduction.
// ---------------------------------------------------------------------------
__global__ __launch_bounds__(256)
void dot_ws_kernel(const float* __restrict__ H, const float* __restrict__ Ws,
                   float* __restrict__ hs, int nnodes)
{
    const int t    = blockIdx.x * blockDim.x + threadIdx.x;
    const int node = t >> 5;
    const int lane = t & 31;
    if (node >= nnodes) return;
    const float* hp = H + (size_t)node * NH + lane * 4;
    const float* wp = Ws + lane * 4;
    float s = hp[0]*wp[0] + hp[1]*wp[1] + hp[2]*wp[2] + hp[3]*wp[3];
    #pragma unroll
    for (int off = 16; off > 0; off >>= 1) s += __shfl_xor(s, off, 32);
    if (lane == 0) hs[node] = s;
}

__global__ void zero_f_kernel(float* p, int n)
{ int i = blockIdx.x * blockDim.x + threadIdx.x; if (i < n) p[i] = 0.0f; }

__global__ void fill_i32_kernel(int* p, int n, int v)
{ int i = blockIdx.x * blockDim.x + threadIdx.x; if (i < n) p[i] = v; }

// deg count: #active in-edges per node (self-loop +1 applied at use sites)
__global__ void deg_kernel(const int* __restrict__ src, const int* __restrict__ dst,
                           float* __restrict__ degc, int ne, int masked)
{
    int e = blockIdx.x * blockDim.x + threadIdx.x;
    if (e >= ne) return;
    int s = src[e];
    if (masked && s < 0) return;
    atomicAdd(&degc[dst[e]], 1.0f);
}

// agg[d] += rsqrt(deg[s]+1)*rsqrt(deg[d]+1) * hs[s]
__global__ void agg_kernel(const int* __restrict__ src, const int* __restrict__ dst,
                           const float* __restrict__ degc, const float* __restrict__ hs,
                           float* __restrict__ agg, int ne, int masked)
{
    int e = blockIdx.x * blockDim.x + threadIdx.x;
    if (e >= ne) return;
    int s = src[e];
    if (masked && s < 0) return;
    int d = dst[e];
    float coef = rsqrtf(degc[s] + 1.0f) * rsqrtf(degc[d] + 1.0f);
    atomicAdd(&agg[d], coef * hs[s]);
}

// score = agg + hs/deg + b ; duplicated into both output slots
__global__ void score_finalize_kernel(const float* __restrict__ agg,
                                      const float* __restrict__ hs,
                                      const float* __restrict__ degc,
                                      const float* __restrict__ bs,
                                      float* __restrict__ o1, float* __restrict__ o2, int n)
{
    int i = blockIdx.x * blockDim.x + threadIdx.x;
    if (i >= n) return;
    float s = agg[i] + hs[i] / (degc[i] + 1.0f) + bs[0];
    o1[i] = s;
    o2[i] = s;
}

// Per-graph rank-based top-k (perm_n) and bottom-k (perm_p); ties -> lower index
// first (matches jax.lax.top_k stability). n <= 1024, k = n/2.
__global__ __launch_bounds__(256)
void topk_kernel(const float* __restrict__ score, int* __restrict__ perm_n,
                 int* __restrict__ perm_p, int n, int k)
{
    __shared__ float sc[1024];
    const int g = blockIdx.x;
    const float* sg = score + (size_t)g * n;
    for (int i = threadIdx.x; i < n; i += 256) sc[i] = sg[i];
    __syncthreads();
    for (int i = threadIdx.x; i < n; i += 256) {
        const float si = sc[i];
        int rt = 0, rb = 0;
        for (int j = 0; j < n; ++j) {
            const float sj = sc[j];
            const bool tie = (sj == si) && (j < i);
            rt += (sj > si) || tie;
            rb += (sj < si) || tie;
        }
        const int gi = g * n + i;
        if (rt < k) perm_n[g * k + rt] = gi;
        if (rb < k) perm_p[g * k + rb] = gi;
    }
}

// mp[perm[t]] = t   (relabel kept nodes)
__global__ void scatter_mp_kernel(const int* __restrict__ perm, int* __restrict__ mp, int m)
{ int t = blockIdx.x * blockDim.x + threadIdx.x; if (t < m) mp[perm[t]] = t; }

// filter_adj: relabel endpoints, mark dropped edges with -1
__global__ void edge_filter_kernel(const int* __restrict__ src, const int* __restrict__ dst,
                                   const int* __restrict__ mp,
                                   int* __restrict__ nsrc, int* __restrict__ ndst,
                                   int ne, int masked)
{
    int e = blockIdx.x * blockDim.x + threadIdx.x;
    if (e >= ne) return;
    int s = src[e];
    if (masked && s < 0) { nsrc[e] = -1; ndst[e] = -1; return; }
    int ms = mp[s], md = mp[dst[e]];
    if (ms >= 0 && md >= 0) { nsrc[e] = ms; ndst[e] = md; }
    else                    { nsrc[e] = -1; ndst[e] = -1; }
}

// readout (gmp || gap) over gathered+gated rows, accumulated (+=) into out[B,256]
__global__ __launch_bounds__(128)
void readout_add_kernel(const float* __restrict__ H, const int* __restrict__ perm,
                        const float* __restrict__ score, float* __restrict__ out, int k)
{
    __shared__ int   pidx[512];
    __shared__ float pts[512];
    const int g = blockIdx.x;
    for (int r = threadIdx.x; r < k; r += 128) {
        int idx = perm[g * k + r];
        pidx[r] = idx;
        pts[r]  = tanhf(score[idx]);
    }
    __syncthreads();
    const int f = threadIdx.x;
    float mx = -3.402823466e38f, sm = 0.0f;
    for (int r = 0; r < k; ++r) {
        float v = H[(size_t)pidx[r] * NH + f] * pts[r];
        mx = fmaxf(mx, v);
        sm += v;
    }
    out[g * 256 + f]       += mx;
    out[g * 256 + 128 + f] += sm / (float)k;
}

// Head MLP: 256 -> 128 (relu) -> 64 (relu) -> 10 -> log_softmax. One block/graph.
__global__ __launch_bounds__(128)
void head_kernel(const float* __restrict__ V,
                 const float* __restrict__ Wl1, const float* __restrict__ bl1,
                 const float* __restrict__ Wl2, const float* __restrict__ bl2,
                 const float* __restrict__ Wl3, const float* __restrict__ bl3,
                 float* __restrict__ out)
{
    __shared__ float v0[256], v1[128], v2[64], lg[10];
    const int g = blockIdx.x, t = threadIdx.x;
    v0[t]       = V[g * 256 + t];
    v0[t + 128] = V[g * 256 + 128 + t];
    __syncthreads();
    float a1 = bl1[t];
    for (int k = 0; k < 256; ++k) a1 += v0[k] * Wl1[k * 128 + t];
    v1[t] = fmaxf(a1, 0.0f);
    __syncthreads();
    if (t < 64) {
        float a2 = bl2[t];
        for (int k = 0; k < 128; ++k) a2 += v1[k] * Wl2[k * 64 + t];
        v2[t] = fmaxf(a2, 0.0f);
    }
    __syncthreads();
    if (t < 10) {
        float a3 = bl3[t];
        for (int k = 0; k < 64; ++k) a3 += v2[k] * Wl3[k * 10 + t];
        lg[t] = a3;
    }
    __syncthreads();
    if (t == 0) {
        float m = lg[0];
        for (int c = 1; c < 10; ++c) m = fmaxf(m, lg[c]);
        float s = 0.0f;
        for (int c = 0; c < 10; ++c) s += expf(lg[c] - m);
        const float lse = m + logf(s);
        for (int c = 0; c < 10; ++c) out[g * 10 + c] = lg[c] - lse;
    }
}

// ---------------------------------------------------------------------------
// Host-side orchestration
// ---------------------------------------------------------------------------
static inline int cdiv(int a, int b) { return (a + b - 1) / b; }

extern "C" void kernel_launch(void* const* d_in, const int* in_sizes, int n_in,
                              void* d_out, int out_size, void* d_ws, size_t ws_size,
                              hipStream_t stream)
{
    (void)in_sizes; (void)n_in; (void)out_size; (void)ws_size;

    const float* x    = (const float*)d_in[0];
    const int*   src0 = (const int*)d_in[1];
    const int*   dst0 = (const int*)d_in[2];
    /* d_in[3] = batch (structural, unused) */
    const float* W1 = (const float*)d_in[4];   const float* b1 = (const float*)d_in[5];
    const float* W2 = (const float*)d_in[6];   const float* b2 = (const float*)d_in[7];
    const float* W3 = (const float*)d_in[8];   const float* b3 = (const float*)d_in[9];
    const float* Ws1 = (const float*)d_in[10]; const float* bs1 = (const float*)d_in[11];
    const float* Ws2 = (const float*)d_in[12]; const float* bs2 = (const float*)d_in[13];
    const float* Ws3 = (const float*)d_in[14]; const float* bs3 = (const float*)d_in[15];
    const float* Wl1 = (const float*)d_in[16]; const float* bl1 = (const float*)d_in[17];
    const float* Wl2 = (const float*)d_in[18]; const float* bl2 = (const float*)d_in[19];
    const float* Wl3 = (const float*)d_in[20]; const float* bl3 = (const float*)d_in[21];

    float* out = (float*)d_out;
    // Output slots (floats): heads, then s1,s1,s2,s2,s3,s3
    float* s1a = out + 5120;
    float* s1b = s1a + NODES1;
    float* s2a = s1b + NODES1;
    float* s2b = s2a + NODES2;
    float* s3a = s2b + NODES2;
    float* s3b = s3a + NODES3;

    // Workspace carve-up
    char* ws = (char*)d_ws;
    auto carve = [&](size_t bytes) -> char* {
        char* p = ws; ws += (bytes + 255) & ~(size_t)255; return p;
    };
    float*  H1     = (float*)carve((size_t)NODES1 * NH * 4);
    float*  H2     = (float*)carve((size_t)NODES2 * NH * 4);
    float*  H3     = (float*)carve((size_t)NODES3 * NH * 4);
    float*  hs     = (float*)carve((size_t)NODES1 * 4);
    float*  degc   = (float*)carve((size_t)NODES1 * 4);
    float*  aggv   = (float*)carve((size_t)NODES1 * 4);
    int*    perm_n = (int*)carve((size_t)Bc * K1c * 4);
    int*    perm_p = (int*)carve((size_t)Bc * K1c * 4);
    int*    mp     = (int*)carve((size_t)NODES1 * 4);
    int*    srcA   = (int*)carve((size_t)Ec * 4);
    int*    dstA   = (int*)carve((size_t)Ec * 4);
    int*    srcB   = (int*)carve((size_t)Ec * 4);
    int*    dstB   = (int*)carve((size_t)Ec * 4);
    float*  xr     = (float*)carve((size_t)Bc * 256 * 4);
    float*  nr     = (float*)carve((size_t)Bc * 256 * 4);
    bf16_t* WT1    = (bf16_t*)carve((size_t)NH * NH * 2);
    bf16_t* WT2    = (bf16_t*)carve((size_t)NH * NH * 2);
    bf16_t* WT3    = (bf16_t*)carve((size_t)NH * NH * 2);

    const int TB = 256;

    // per-launch init: readout accumulators and transposed bf16 weights
    zero_f_kernel<<<cdiv(Bc * 256, TB), TB, 0, stream>>>(xr, Bc * 256);
    zero_f_kernel<<<cdiv(Bc * 256, TB), TB, 0, stream>>>(nr, Bc * 256);
    transpose_w_bf16_kernel<<<NH * NH / TB, TB, 0, stream>>>(W1, WT1);
    transpose_w_bf16_kernel<<<NH * NH / TB, TB, 0, stream>>>(W2, WT2);
    transpose_w_bf16_kernel<<<NH * NH / TB, TB, 0, stream>>>(W3, WT3);

    // ---------------- level 0 conv: H1 = relu(x @ W1 + b1) ----------------
    gemm_relu_wmma<<<NODES1 / 64, 256, 0, stream>>>(x, WT1, b1, H1, nullptr, nullptr);

    // ---------------- level 1 ----------------
    dot_ws_kernel<<<cdiv(NODES1 * 32, TB), TB, 0, stream>>>(H1, Ws1, hs, NODES1);
    zero_f_kernel<<<cdiv(NODES1, TB), TB, 0, stream>>>(degc, NODES1);
    zero_f_kernel<<<cdiv(NODES1, TB), TB, 0, stream>>>(aggv, NODES1);
    deg_kernel<<<cdiv(Ec, TB), TB, 0, stream>>>(src0, dst0, degc, Ec, 0);
    agg_kernel<<<cdiv(Ec, TB), TB, 0, stream>>>(src0, dst0, degc, hs, aggv, Ec, 0);
    score_finalize_kernel<<<cdiv(NODES1, TB), TB, 0, stream>>>(aggv, hs, degc, bs1, s1a, s1b, NODES1);
    topk_kernel<<<Bc, 256, 0, stream>>>(s1a, perm_n, perm_p, Nc, K1c);
    readout_add_kernel<<<Bc, 128, 0, stream>>>(H1, perm_p, s1a, xr, K1c);
    readout_add_kernel<<<Bc, 128, 0, stream>>>(H1, perm_n, s1a, nr, K1c);
    // H2 = relu( (tanh(s1)*H1)[perm_p] @ W2 + b2 )  -- xp1 never materialized
    gemm_relu_wmma<<<NODES2 / 64, 256, 0, stream>>>(H1, WT2, b2, H2, perm_p, s1a);
    fill_i32_kernel<<<cdiv(NODES1, TB), TB, 0, stream>>>(mp, NODES1, -1);
    scatter_mp_kernel<<<cdiv(NODES2, TB), TB, 0, stream>>>(perm_p, mp, NODES2);
    edge_filter_kernel<<<cdiv(Ec, TB), TB, 0, stream>>>(src0, dst0, mp, srcA, dstA, Ec, 0);

    // ---------------- level 2 ----------------
    dot_ws_kernel<<<cdiv(NODES2 * 32, TB), TB, 0, stream>>>(H2, Ws2, hs, NODES2);
    zero_f_kernel<<<cdiv(NODES2, TB), TB, 0, stream>>>(degc, NODES2);
    zero_f_kernel<<<cdiv(NODES2, TB), TB, 0, stream>>>(aggv, NODES2);
    deg_kernel<<<cdiv(Ec, TB), TB, 0, stream>>>(srcA, dstA, degc, Ec, 1);
    agg_kernel<<<cdiv(Ec, TB), TB, 0, stream>>>(srcA, dstA, degc, hs, aggv, Ec, 1);
    score_finalize_kernel<<<cdiv(NODES2, TB), TB, 0, stream>>>(aggv, hs, degc, bs2, s2a, s2b, NODES2);
    topk_kernel<<<Bc, 256, 0, stream>>>(s2a, perm_n, perm_p, K1c, K2c);
    readout_add_kernel<<<Bc, 128, 0, stream>>>(H2, perm_p, s2a, xr, K2c);
    readout_add_kernel<<<Bc, 128, 0, stream>>>(H2, perm_n, s2a, nr, K2c);
    gemm_relu_wmma<<<NODES3 / 64, 256, 0, stream>>>(H2, WT3, b3, H3, perm_p, s2a);
    fill_i32_kernel<<<cdiv(NODES2, TB), TB, 0, stream>>>(mp, NODES2, -1);
    scatter_mp_kernel<<<cdiv(NODES3, TB), TB, 0, stream>>>(perm_p, mp, NODES3);
    edge_filter_kernel<<<cdiv(Ec, TB), TB, 0, stream>>>(srcA, dstA, mp, srcB, dstB, Ec, 1);

    // ---------------- level 3 ----------------
    dot_ws_kernel<<<cdiv(NODES3 * 32, TB), TB, 0, stream>>>(H3, Ws3, hs, NODES3);
    zero_f_kernel<<<cdiv(NODES3, TB), TB, 0, stream>>>(degc, NODES3);
    zero_f_kernel<<<cdiv(NODES3, TB), TB, 0, stream>>>(aggv, NODES3);
    deg_kernel<<<cdiv(Ec, TB), TB, 0, stream>>>(srcB, dstB, degc, Ec, 1);
    agg_kernel<<<cdiv(Ec, TB), TB, 0, stream>>>(srcB, dstB, degc, hs, aggv, Ec, 1);
    score_finalize_kernel<<<cdiv(NODES3, TB), TB, 0, stream>>>(aggv, hs, degc, bs3, s3a, s3b, NODES3);
    topk_kernel<<<Bc, 256, 0, stream>>>(s3a, perm_n, perm_p, K2c, K3c);
    readout_add_kernel<<<Bc, 128, 0, stream>>>(H3, perm_p, s3a, xr, K3c);
    readout_add_kernel<<<Bc, 128, 0, stream>>>(H3, perm_n, s3a, nr, K3c);

    // ---------------- heads ----------------
    head_kernel<<<Bc, 128, 0, stream>>>(xr, Wl1, bl1, Wl2, bl2, Wl3, bl3, out);
    head_kernel<<<Bc, 128, 0, stream>>>(nr, Wl1, bl1, Wl2, bl2, Wl3, bl3, out + 2560);
}
